// MHA_60146722013367
// MI455X (gfx1250) — matrix operand order repaired and checked by reading, not compiled, not run on hardware
//
#include <hip/hip_runtime.h>

// ---------------------------------------------------------------------------
// MHA block for MI455X (gfx1250, wave32, WMMA).
//   x      : [2, 2048, 1024] f32
//   w_qkv  : [3072, 1024]    f32   (row = output feature, col = input feature)
//   w_out  : [1024, 1024]    f32
//   out    : [2, 2048, 1024] f32
// Strategy: convert to f16 once, run every GEMM stage on
// v_wmma_f32_16x16x32_f16 (f32 accumulate), flash-attention inner loop.
// Workspace requirement: 48 MiB (7 f16 tensors, see kernel_launch).
// ---------------------------------------------------------------------------

typedef _Float16 h16;
typedef __attribute__((ext_vector_type(16))) _Float16 v16h;
typedef __attribute__((ext_vector_type(8)))  _Float16 v8h;
typedef __attribute__((ext_vector_type(8)))  float    v8f;

#define HIDDEN 1024
#define SEQ    2048
#define BATCH  2
#define HEADS  16
#define HD     64
#define MROWS  (BATCH * SEQ)   // 4096
#define SCALE  0.125f          // 64^-0.5

// ---- fragment loaders (contiguous k order k=16*hi+s, same for A and B) ----------

// A-matrix 16x32 (MxK): lane m=(lane&15), hi=(lane>>4) holds
// K = [16*hi .. 16*hi+15] of row m.
__device__ __forceinline__ v16h ld_a16(const h16* __restrict__ rowp, int hi) {
  v8h lo = *(const v8h*)(rowp + 16 * hi);
  v8h hi8 = *(const v8h*)(rowp + 16 * hi + 8);
  return __builtin_shufflevector(lo, hi8, 0, 1, 2, 3, 4, 5, 6, 7,
                                 8, 9, 10, 11, 12, 13, 14, 15);
}

// B-matrix 32x16 (KxN): lane n=(lane&15), hi=(lane>>4) holds
// K = [16*hi .. 16*hi+15] of column n (p must point at that run).
__device__ __forceinline__ v16h ld_b16(const h16* __restrict__ p) {
  v8h lo = *(const v8h*)(p);
  v8h hi8 = *(const v8h*)(p + 8);
  return __builtin_shufflevector(lo, hi8, 0, 1, 2, 3, 4, 5, 6, 7,
                                 8, 9, 10, 11, 12, 13, 14, 15);
}

__device__ __forceinline__ v8f wmma16(v16h a, v16h b, v8f c) {
  return __builtin_amdgcn_wmma_f32_16x16x32_f16(
      /*neg_a=*/false, a, /*neg_b=*/false, b,
      /*c_mod=*/(short)0, c, /*reuse_a=*/false, /*reuse_b=*/false);
}

// ---- fp32 -> f16 conversion ----------------------------------------------

__global__ __launch_bounds__(256) void cvt_f32_f16_kernel(
    const float* __restrict__ in, h16* __restrict__ out, int n) {
  int i = blockIdx.x * 256 + threadIdx.x;
  if (i < n) out[i] = (h16)in[i];
}

// ---- QKV projection: C[M=4096, E=3072] = X[M,1024] * Wqkv[E,1024]^T -------
// Epilogue scatters: Q,K -> [b,h,s,hd] f16 ; V -> transposed [b,h,hd,s] f16.

__device__ __forceinline__ void store_qkv_frag(
    const v8f& c, int mb, int nb, int hi, int ln,
    h16* __restrict__ Qo, h16* __restrict__ Ko, h16* __restrict__ Vt) {
#pragma unroll
  for (int r = 0; r < 8; ++r) {
    int row = mb + r + 8 * hi;      // b*SEQ + s
    int e   = nb + ln;              // 0..3071
    int which = e >> 10;
    int f = e & 1023;
    int h = f >> 6, d = f & 63;
    int bb = row >> 11, s = row & (SEQ - 1);
    int bh = bb * HEADS + h;
    h16 v = (h16)c[r];
    if (which == 0)      Qo[((size_t)bh * SEQ + s) * HD + d] = v;
    else if (which == 1) Ko[((size_t)bh * SEQ + s) * HD + d] = v;
    else                 Vt[((size_t)bh * HD + d) * SEQ + s] = v;
  }
}

__global__ __launch_bounds__(256) void qkv_gemm_kernel(
    const h16* __restrict__ X, const h16* __restrict__ W,
    h16* __restrict__ Qo, h16* __restrict__ Ko, h16* __restrict__ Vt) {
  const int lane = threadIdx.x & 31;
  const int wave = threadIdx.x >> 5;           // 0..7
  const int hi = lane >> 4, ln = lane & 15;
  const int m0 = blockIdx.y * 64 + (wave >> 2) * 32;   // M block 64
  const int n0 = blockIdx.x * 128 + (wave & 3) * 32;   // N block 128

  v8f c00 = {}, c01 = {}, c10 = {}, c11 = {};
  const h16* a0p = X + (size_t)(m0 + ln) * HIDDEN;
  const h16* a1p = X + (size_t)(m0 + 16 + ln) * HIDDEN;
  const h16* b0p = W + (size_t)(n0 + ln) * HIDDEN + 16 * hi;
  const h16* b1p = W + (size_t)(n0 + 16 + ln) * HIDDEN + 16 * hi;

  for (int kk = 0; kk < HIDDEN; kk += 32) {
    v16h a0 = ld_a16(a0p + kk, hi);
    v16h a1 = ld_a16(a1p + kk, hi);
    v16h b0 = ld_b16(b0p + kk);
    v16h b1 = ld_b16(b1p + kk);
    c00 = wmma16(a0, b0, c00);
    c01 = wmma16(a0, b1, c01);
    c10 = wmma16(a1, b0, c10);
    c11 = wmma16(a1, b1, c11);
  }
  store_qkv_frag(c00, m0,      n0,      hi, ln, Qo, Ko, Vt);
  store_qkv_frag(c01, m0,      n0 + 16, hi, ln, Qo, Ko, Vt);
  store_qkv_frag(c10, m0 + 16, n0,      hi, ln, Qo, Ko, Vt);
  store_qkv_frag(c11, m0 + 16, n0 + 16, hi, ln, Qo, Ko, Vt);
}

// ---- Flash attention: one wave per 16-query tile per (b,h) ----------------
// Q,K: [b,h,s,64] f16 ; Vt: [b,h,64,s] f16 ; out: [b,s,1024] f16.

__global__ __launch_bounds__(128) void attn_fa_kernel(
    const h16* __restrict__ Qm, const h16* __restrict__ Km,
    const h16* __restrict__ Vt, h16* __restrict__ Om) {
  __shared__ alignas(16) h16 lds_p[4][16 * 32];   // per-wave 16x32 prob tile

  const int lane = threadIdx.x & 31;
  const int wave = threadIdx.x >> 5;              // 0..3
  const int hi = lane >> 4, ln = lane & 15;
  const int bh = blockIdx.x;                      // 0..31
  const int b = bh >> 4, h = bh & 15;
  const int qt = blockIdx.y * 4 + wave;           // 0..127
  const int q0 = qt * 16;

  const h16* Qb = Qm + (size_t)bh * SEQ * HD;
  const h16* Kb = Km + (size_t)bh * SEQ * HD;
  const h16* Vb = Vt + (size_t)bh * HD * SEQ;

  // Q fragments for the whole hd=64 reduction (two K=32 steps), kept resident.
  v16h aq0 = ld_a16(Qb + (size_t)(q0 + ln) * HD + 0, hi);
  v16h aq1 = ld_a16(Qb + (size_t)(q0 + ln) * HD + 32, hi);

  v8f o0 = {}, o1 = {}, o2 = {}, o3 = {};        // 16 queries x 64 dims
  float mrow[8], lrow[8];
#pragma unroll
  for (int r = 0; r < 8; ++r) { mrow[r] = -__builtin_inff(); lrow[r] = 0.0f; }

  h16* lp = &lds_p[wave][0];

  for (int kb = 0; kb < SEQ; kb += 32) {
    // scores: S[16 x 32] = Q * K^T, split as two 16x16 WMMA outputs
    v16h bk0a = ld_b16(Kb + (size_t)(kb + ln) * HD + 16 * hi);
    v16h bk0b = ld_b16(Kb + (size_t)(kb + ln) * HD + 32 + 16 * hi);
    v16h bk1a = ld_b16(Kb + (size_t)(kb + 16 + ln) * HD + 16 * hi);
    v16h bk1b = ld_b16(Kb + (size_t)(kb + 16 + ln) * HD + 32 + 16 * hi);
    v8f s0 = {}, s1 = {};
    s0 = wmma16(aq0, bk0a, s0);
    s0 = wmma16(aq1, bk0b, s0);
    s1 = wmma16(aq0, bk1a, s1);
    s1 = wmma16(aq1, bk1b, s1);

    // online softmax; row r of a C-frag lives in lanes {0..15}+16*hi
    float sf[8];
#pragma unroll
    for (int r = 0; r < 8; ++r) {
      float x0 = s0[r] * SCALE;
      float x1 = s1[r] * SCALE;
      float t = fmaxf(x0, x1);
      t = fmaxf(t, __shfl_xor(t, 1));
      t = fmaxf(t, __shfl_xor(t, 2));
      t = fmaxf(t, __shfl_xor(t, 4));
      t = fmaxf(t, __shfl_xor(t, 8));
      float nm = fmaxf(mrow[r], t);
      sf[r] = __expf(mrow[r] - nm);
      mrow[r] = nm;
      float p0 = __expf(x0 - nm);
      float p1 = __expf(x1 - nm);
      lrow[r] = lrow[r] * sf[r] + p0 + p1;      // per-lane partial row sum
      // stage probabilities for the A-fragment re-layout
      lp[(r + 8 * hi) * 32 + ln]      = (h16)p0;
      lp[(r + 8 * hi) * 32 + 16 + ln] = (h16)p1;
    }
    asm volatile("s_wait_dscnt 0x0" ::: "memory");   // same-wave DS ordering

    // P as A-fragment (16 queries x 32 keys)
    v16h pa;
    {
      v8h plo = *(const v8h*)(lp + ln * 32 + 16 * hi);
      v8h phi = *(const v8h*)(lp + ln * 32 + 16 * hi + 8);
      pa = __builtin_shufflevector(plo, phi, 0, 1, 2, 3, 4, 5, 6, 7,
                                   8, 9, 10, 11, 12, 13, 14, 15);
    }

    // rescale accumulators, then O += P * V
#pragma unroll
    for (int r = 0; r < 8; ++r) {
      o0[r] *= sf[r]; o1[r] *= sf[r]; o2[r] *= sf[r]; o3[r] *= sf[r];
    }
    v16h bv0 = ld_b16(Vb + (size_t)(0 * 16 + ln) * SEQ + kb + 16 * hi);
    v16h bv1 = ld_b16(Vb + (size_t)(1 * 16 + ln) * SEQ + kb + 16 * hi);
    v16h bv2 = ld_b16(Vb + (size_t)(2 * 16 + ln) * SEQ + kb + 16 * hi);
    v16h bv3 = ld_b16(Vb + (size_t)(3 * 16 + ln) * SEQ + kb + 16 * hi);
    o0 = wmma16(pa, bv0, o0);
    o1 = wmma16(pa, bv1, o1);
    o2 = wmma16(pa, bv2, o2);
    o3 = wmma16(pa, bv3, o3);
  }

  // finalize: reduce row sums across the 16-lane half, normalize, store f16
#pragma unroll
  for (int r = 0; r < 8; ++r) {
    float t = lrow[r];
    t += __shfl_xor(t, 1);
    t += __shfl_xor(t, 2);
    t += __shfl_xor(t, 4);
    t += __shfl_xor(t, 8);
    float inv = 1.0f / t;
    int s = q0 + r + 8 * hi;
    h16* orow = Om + ((size_t)(b * SEQ + s)) * HIDDEN + h * HD;
    orow[0 * 16 + ln] = (h16)(o0[r] * inv);
    orow[1 * 16 + ln] = (h16)(o1[r] * inv);
    orow[2 * 16 + ln] = (h16)(o2[r] * inv);
    orow[3 * 16 + ln] = (h16)(o3[r] * inv);
  }
}

// ---- output projection: out[M=4096,1024] = A[M,1024] * Wout[1024,1024]^T --

__global__ __launch_bounds__(256) void out_gemm_kernel(
    const h16* __restrict__ A, const h16* __restrict__ W,
    float* __restrict__ C) {
  const int lane = threadIdx.x & 31;
  const int wave = threadIdx.x >> 5;
  const int hi = lane >> 4, ln = lane & 15;
  const int m0 = blockIdx.y * 64 + (wave >> 2) * 32;
  const int n0 = blockIdx.x * 128 + (wave & 3) * 32;

  v8f c00 = {}, c01 = {}, c10 = {}, c11 = {};
  const h16* a0p = A + (size_t)(m0 + ln) * HIDDEN;
  const h16* a1p = A + (size_t)(m0 + 16 + ln) * HIDDEN;
  const h16* b0p = W + (size_t)(n0 + ln) * HIDDEN + 16 * hi;
  const h16* b1p = W + (size_t)(n0 + 16 + ln) * HIDDEN + 16 * hi;

  for (int kk = 0; kk < HIDDEN; kk += 32) {
    v16h a0 = ld_a16(a0p + kk, hi);
    v16h a1 = ld_a16(a1p + kk, hi);
    v16h b0 = ld_b16(b0p + kk);
    v16h b1 = ld_b16(b1p + kk);
    c00 = wmma16(a0, b0, c00);
    c01 = wmma16(a0, b1, c01);
    c10 = wmma16(a1, b0, c10);
    c11 = wmma16(a1, b1, c11);
  }
#pragma unroll
  for (int r = 0; r < 8; ++r) {
    int row0 = m0 + r + 8 * hi;
    int row1 = m0 + 16 + r + 8 * hi;
    C[(size_t)row0 * HIDDEN + n0 + ln]      = c00[r];
    C[(size_t)row0 * HIDDEN + n0 + 16 + ln] = c01[r];
    C[(size_t)row1 * HIDDEN + n0 + ln]      = c10[r];
    C[(size_t)row1 * HIDDEN + n0 + 16 + ln] = c11[r];
  }
}

// ---------------------------------------------------------------------------

extern "C" void kernel_launch(void* const* d_in, const int* in_sizes, int n_in,
                              void* d_out, int out_size, void* d_ws, size_t ws_size,
                              hipStream_t stream) {
  const float* x     = (const float*)d_in[0];   // [2,2048,1024]
  const float* w_qkv = (const float*)d_in[1];   // [3072,1024]
  const float* w_out = (const float*)d_in[2];   // [1024,1024]

  // workspace layout (f16), total 48 MiB
  char* ws = (char*)d_ws;
  h16* xh  = (h16*)ws; ws += (size_t)MROWS * HIDDEN * 2;        // 8 MiB
  h16* wqh = (h16*)ws; ws += (size_t)3 * HIDDEN * HIDDEN * 2;   // 6 MiB
  h16* woh = (h16*)ws; ws += (size_t)HIDDEN * HIDDEN * 2;       // 2 MiB
  h16* q   = (h16*)ws; ws += (size_t)MROWS * HIDDEN * 2;        // 8 MiB
  h16* k   = (h16*)ws; ws += (size_t)MROWS * HIDDEN * 2;        // 8 MiB
  h16* vt  = (h16*)ws; ws += (size_t)MROWS * HIDDEN * 2;        // 8 MiB
  h16* ao  = (h16*)ws; ws += (size_t)MROWS * HIDDEN * 2;        // 8 MiB

  int nx = MROWS * HIDDEN;
  int nq = 3 * HIDDEN * HIDDEN;
  int no = HIDDEN * HIDDEN;
  cvt_f32_f16_kernel<<<(nx + 255) / 256, 256, 0, stream>>>(x, xh, nx);
  cvt_f32_f16_kernel<<<(nq + 255) / 256, 256, 0, stream>>>(w_qkv, wqh, nq);
  cvt_f32_f16_kernel<<<(no + 255) / 256, 256, 0, stream>>>(w_out, woh, no);

  qkv_gemm_kernel<<<dim3(3 * HIDDEN / 128, MROWS / 64), 256, 0, stream>>>(
      xh, wqh, q, k, vt);

  attn_fa_kernel<<<dim3(BATCH * HEADS, SEQ / 64), 128, 0, stream>>>(
      q, k, vt, ao);

  out_gemm_kernel<<<dim3(HIDDEN / 128, MROWS / 64), 256, 0, stream>>>(
      ao, woh, (float*)d_out);
}
